// ReasoningAwareAttention_76235669504433
// MI455X (gfx1250) — matrix operand order repaired and checked by reading, not compiled, not run on hardware
//
#include <hip/hip_runtime.h>

// ---------------------------------------------------------------------------
// Reasoning-aware attention for MI455X (gfx1250), wave32 + WMMA bf16 path.
// Fused scores -> online softmax -> attn write (NT) -> context kernel.
// ---------------------------------------------------------------------------

typedef __bf16 bf16;
typedef __attribute__((ext_vector_type(16))) __bf16 v16bf;
typedef __attribute__((ext_vector_type(8)))  __bf16 bf16x8;
typedef __attribute__((ext_vector_type(8)))  float  v8f;

#define DEV __device__ __forceinline__
#define WMMA_BF16(A_, B_, C_) \
  __builtin_amdgcn_wmma_f32_16x16x32_bf16(false, (A_), false, (B_), (short)0, (C_), false, false)

constexpr int BB = 2;            // batch
constexpr int SS = 2048;         // sequence
constexpr int DD = 1024;         // d_model
constexpr int HH = 16;           // heads
constexpr int DK = 64;           // head dim
constexpr int MM = BB * SS;      // 4096 rows
constexpr float LN_EPS = 1e-5f;

// --------------------------- WMMA fragment loaders -------------------------
// A fragment: 16x32 bf16 tile, A points at tile origin [row0, k0], lda elems.
// lane 0..15 : row = lane,    K = {0..7, 16..23}
// lane 16..31: row = lane-16, K = {8..15, 24..31}
DEV v16bf load_a_bf16(const bf16* A, int lda, int lane) {
  const int row = lane & 15;
  const int kb  = (lane >> 4) * 8;
  const bf16* p = A + row * lda;
  bf16x8 lo = *(const bf16x8*)(p + kb);
  bf16x8 hi = *(const bf16x8*)(p + 16 + kb);
  v16bf a;
#pragma unroll
  for (int j = 0; j < 8; ++j) { a[j] = lo[j]; a[8 + j] = hi[j]; }
  return a;
}

// B fragment: 32x16 bf16 tile B[k,n], stored as rows: Bm + n*ldb + k.
// lane 0..15 : col = lane,    K = {0..7, 16..23} ; lane 16..31: col = lane-16, K = {8..15, 24..31}
DEV v16bf load_b_bf16(const bf16* Bm, int ldb, int lane) {
  const int col = lane & 15;
  const int kh  = (lane >> 4) * 8;
  const bf16* p = Bm + col * ldb + kh;
  bf16x8 lo = *(const bf16x8*)(p);
  bf16x8 hi = *(const bf16x8*)(p + 16);
  v16bf b;
#pragma unroll
  for (int j = 0; j < 8; ++j) { b[j] = lo[j]; b[8 + j] = hi[j]; }
  return b;
}

// A fragment from per-wave LDS staging (fp32 probabilities -> bf16).
DEV v16bf load_a_lds(const float (&st)[16][72], int kc, int lane) {
  const int row = lane & 15;
  const int kb  = (lane >> 4) * 8;
  v16bf a;
#pragma unroll
  for (int j = 0; j < 8; ++j) {
    a[j]     = (bf16)st[row][kc + kb + j];
    a[8 + j] = (bf16)st[row][kc + 16 + kb + j];
  }
  return a;
}

// Core: one wave computes a 16(M) x 64(N) tile, K assumed multiple of 64.
// Ping-pong unroll-by-2: each fragment buffer is written directly by loads
// (no rotating copies -> no v_mov storm), loads for one half are in flight
// while the other half's WMMAs execute. Final prefetch wraps to 0 (valid,
// redundant) to keep loads unconditional.
DEV void gemm_tile_bf16(const bf16* A, int lda, const bf16* Bm, int ldb,
                        int K, int lane, v8f acc[4]) {
  v16bf a0 = load_a_bf16(A, lda, lane);
  v16bf b0[4];
#pragma unroll
  for (int t = 0; t < 4; ++t) b0[t] = load_b_bf16(Bm + t * 16 * ldb, ldb, lane);
  for (int k0 = 0; k0 < K; k0 += 64) {
    // prefetch odd half (always in range: K % 64 == 0)
    v16bf a1 = load_a_bf16(A + k0 + 32, lda, lane);
    v16bf b1[4];
#pragma unroll
    for (int t = 0; t < 4; ++t)
      b1[t] = load_b_bf16(Bm + t * 16 * ldb + k0 + 32, ldb, lane);
#pragma unroll
    for (int t = 0; t < 4; ++t) acc[t] = WMMA_BF16(a0, b0[t], acc[t]);
    // prefetch next even half (wraps on last iteration; harmless)
    const int kn = (k0 + 64 < K) ? (k0 + 64) : 0;
    a0 = load_a_bf16(A + kn, lda, lane);
#pragma unroll
    for (int t = 0; t < 4; ++t)
      b0[t] = load_b_bf16(Bm + t * 16 * ldb + kn, ldb, lane);
#pragma unroll
    for (int t = 0; t < 4; ++t) acc[t] = WMMA_BF16(a1, b1[t], acc[t]);
  }
}

// ------------------------------- Kernels -----------------------------------

__global__ void cvt_kernel(const float* __restrict__ src, bf16* __restrict__ dst, int n) {
  int i = blockIdx.x * blockDim.x + threadIdx.x;
  if (i < n) dst[i] = (bf16)src[i];
}

// Q/K projection: out[m,n] = sum_k A[m,k] * W[n,k] + bias[n], bf16 output.
__global__ void gemm_proj_kernel(const bf16* __restrict__ A, const bf16* __restrict__ W,
                                 const float* __restrict__ bias, bf16* __restrict__ out) {
  const int wave = (blockIdx.x * blockDim.x + threadIdx.x) >> 5;
  const int lane = threadIdx.x & 31;
  const int ntn = DD / 64;
  const int m0 = (wave / ntn) * 16;
  const int n0 = (wave % ntn) * 64;
  v8f acc[4] = {{}, {}, {}, {}};
  gemm_tile_bf16(A + (size_t)m0 * DD, DD, W + (size_t)n0 * DD, DD, DD, lane, acc);
  const int col = lane & 15, grp = lane >> 4;
#pragma unroll
  for (int t = 0; t < 4; ++t)
#pragma unroll
    for (int r = 0; r < 8; ++r) {
      const int m = m0 + r + 8 * grp;
      const int n = n0 + t * 16 + col;
      out[(size_t)m * DD + n] = (bf16)(acc[t][r] + bias[n]);
    }
}

// V projection with transposed store: Vt[b, channel, token] (bf16), so the
// context B-fragments become contiguous 16B loads along the token axis.
__global__ void gemm_v_kernel(const bf16* __restrict__ A, const bf16* __restrict__ W,
                              const float* __restrict__ bias, bf16* __restrict__ Vt) {
  const int wave = (blockIdx.x * blockDim.x + threadIdx.x) >> 5;
  const int lane = threadIdx.x & 31;
  const int ntn = DD / 64;
  const int m0 = (wave / ntn) * 16;
  const int n0 = (wave % ntn) * 64;
  v8f acc[4] = {{}, {}, {}, {}};
  gemm_tile_bf16(A + (size_t)m0 * DD, DD, W + (size_t)n0 * DD, DD, DD, lane, acc);
  const int col = lane & 15, grp = lane >> 4;
#pragma unroll
  for (int t = 0; t < 4; ++t)
#pragma unroll
    for (int r = 0; r < 8; ++r) {
      const int m = m0 + r + 8 * grp;
      const int b = m >> 11, s = m & (SS - 1);
      const int n = n0 + t * 16 + col;
      Vt[((size_t)b * DD + n) * SS + s] = (bf16)(acc[t][r] + bias[n]);
    }
}

// R = reasoning_state @ Wr^T + br  (tiny)
__global__ void rproj_kernel(const float* __restrict__ rs, const float* __restrict__ Wr,
                             const float* __restrict__ br, float* __restrict__ R) {
  int idx = blockIdx.x * blockDim.x + threadIdx.x;
  if (idx >= BB * DD) return;
  const int b = idx >> 10, n = idx & (DD - 1);
  const float* a = rs + b * DD;
  const float* w = Wr + (size_t)n * DD;
  float s = 0.f;
  for (int k = 0; k < DD; ++k) s += a[k] * w[k];
  R[idx] = s + br[n];
}

// qr[b,h,s] = Q[b,s,h*64:+64] . R[b,h*64:+64]
__global__ void qr_kernel(const bf16* __restrict__ Qb, const float* __restrict__ R,
                          float* __restrict__ qr) {
  int idx = blockIdx.x * blockDim.x + threadIdx.x;
  if (idx >= BB * HH * SS) return;
  const int b = idx / (HH * SS);
  const int h = (idx / SS) % HH;
  const int s = idx & (SS - 1);
  const bf16*  q = Qb + ((size_t)(b * SS + s)) * DD + h * DK;
  const float* r = R + b * DD + h * DK;
  float acc = 0.f;
#pragma unroll 4
  for (int d = 0; d < DK; ++d) acc += (float)q[d] * r[d];
  qr[((size_t)b * HH + h) * SS + s] = acc;
}

// ---------------------------------------------------------------------------
// Fused attention: per wave, one (b,h) x 16-query-row block over all 2048 keys.
//   pass 1: scores = QK^T/8 + qr via WMMA; online row max/sum (shfl reduce).
//   pass 2: recompute scores, normalize, NT-store attn to d_out, transpose
//           probs through per-wave LDS, accumulate context with WMMA vs Vt.
// attn HBM traffic collapses to the single mandatory 537MB write.
// ---------------------------------------------------------------------------
__global__ void fused_attn_kernel(const bf16* __restrict__ Qb, const bf16* __restrict__ Kb,
                                  const bf16* __restrict__ Vt, const float* __restrict__ qr,
                                  float* __restrict__ attn, bf16* __restrict__ ctx) {
  __shared__ float stage[8][16][72];   // per-wave 16x64 prob tile (+pad), 36.9KB

  const int widx = threadIdx.x >> 5;
  const int lane = threadIdx.x & 31;
  const int unit = blockIdx.x * 8 + widx;        // 4096 units total
  const int bh = unit >> 7;                      // 128 row-blocks per (b,h)
  const int m0 = (unit & 127) * 16;
  const int b = bh >> 4, h = bh & 15;
  const int col = lane & 15, grp = lane >> 4;

  const bf16* Qtile = Qb + ((size_t)(b * SS + m0)) * DD + h * DK;  // [16 x 64]
  const bf16* Kbase = Kb + ((size_t)b * SS) * DD + h * DK;         // [t, k] ldb=DD
  const bf16* Vbase = Vt + ((size_t)b * DD + h * DK) * SS;         // [d, t] ldb=SS
  float (&st)[16][72] = stage[widx];

  // Q fragments are invariant across the whole key loop: load once.
  const v16bf aq0 = load_a_bf16(Qtile, DD, lane);
  const v16bf aq1 = load_a_bf16(Qtile + 32, DD, lane);

  // per-row reasoning bias (softmax-invariant, kept for faithfulness)
  float qrv[8];
#pragma unroll
  for (int r = 0; r < 8; ++r) qrv[r] = qr[(size_t)bh * SS + m0 + r + 8 * grp];

  const float scale = 0.125f;   // 1/sqrt(64)
  float Mrow[8], Lrow[8];
#pragma unroll
  for (int r = 0; r < 8; ++r) { Mrow[r] = -3.4e38f; Lrow[r] = 0.f; }

  // ---------------- pass 1: online max / sum over 32 key chunks -------------
  for (int c = 0; c < SS / 64; ++c) {
    const int tbase = c * 64;
    v8f sacc[4] = {{}, {}, {}, {}};
#pragma unroll
    for (int ks = 0; ks < 2; ++ks) {
      const v16bf a = ks ? aq1 : aq0;
#pragma unroll
      for (int t = 0; t < 4; ++t) {
        v16bf bf = load_b_bf16(Kbase + (size_t)(tbase + t * 16) * DD + ks * 32, DD, lane);
        sacc[t] = WMMA_BF16(a, bf, sacc[t]);
      }
    }
#pragma unroll
    for (int r = 0; r < 8; ++r) {
      float cm = -3.4e38f;
#pragma unroll
      for (int t = 0; t < 4; ++t) cm = fmaxf(cm, sacc[t][r] * scale + qrv[r]);
#pragma unroll
      for (int mk = 1; mk < 16; mk <<= 1) cm = fmaxf(cm, __shfl_xor(cm, mk));
      const float Mn = fmaxf(Mrow[r], cm);
      float cs = 0.f;
#pragma unroll
      for (int t = 0; t < 4; ++t) cs += __expf(sacc[t][r] * scale + qrv[r] - Mn);
#pragma unroll
      for (int mk = 1; mk < 16; mk <<= 1) cs += __shfl_xor(cs, mk);
      Lrow[r] = Lrow[r] * __expf(Mrow[r] - Mn) + cs;
      Mrow[r] = Mn;
    }
  }
  float invL[8];
#pragma unroll
  for (int r = 0; r < 8; ++r) invL[r] = 1.0f / Lrow[r];

  // ------------- pass 2: normalize, write attn (NT), context WMMA -----------
  v8f cacc[4] = {{}, {}, {}, {}};
  for (int c = 0; c < SS / 64; ++c) {
    const int tbase = c * 64;
    v8f sacc[4] = {{}, {}, {}, {}};
#pragma unroll
    for (int ks = 0; ks < 2; ++ks) {
      const v16bf a = ks ? aq1 : aq0;
#pragma unroll
      for (int t = 0; t < 4; ++t) {
        v16bf bf = load_b_bf16(Kbase + (size_t)(tbase + t * 16) * DD + ks * 32, DD, lane);
        sacc[t] = WMMA_BF16(a, bf, sacc[t]);
      }
    }
    // normalize, stream to d_out, stage transpose in LDS
#pragma unroll
    for (int r = 0; r < 8; ++r) {
      const int m = m0 + r + 8 * grp;
      float* orow = attn + ((size_t)bh * SS + m) * SS + tbase;
#pragma unroll
      for (int t = 0; t < 4; ++t) {
        const float p = __expf(sacc[t][r] * scale + qrv[r] - Mrow[r]) * invL[r];
        __builtin_nontemporal_store(p, orow + t * 16 + col);
        st[r + 8 * grp][t * 16 + col] = p;
      }
    }
    __syncthreads();
    // context accumulate: A = probs (LDS, ->bf16), B = Vt rows (contiguous)
#pragma unroll
    for (int ks = 0; ks < 2; ++ks) {
      const v16bf a = load_a_lds(st, ks * 32, lane);
#pragma unroll
      for (int t = 0; t < 4; ++t) {
        v16bf bf = load_b_bf16(Vbase + (size_t)(t * 16) * SS + tbase + ks * 32, SS, lane);
        cacc[t] = WMMA_BF16(a, bf, cacc[t]);
      }
    }
    __syncthreads();
  }

  // write context tile (bf16)
#pragma unroll
  for (int t = 0; t < 4; ++t)
#pragma unroll
    for (int r = 0; r < 8; ++r) {
      const int m = m0 + r + 8 * grp;
      const int n = t * 16 + col;
      ctx[((size_t)(b * SS + m)) * DD + h * DK + n] = (bf16)cacc[t][r];
    }
}

// out = ctx @ Wo^T + bo + x  (fp32 pre-LN into ws)
__global__ void outproj_kernel(const bf16* __restrict__ ctx, const bf16* __restrict__ Wo,
                               const float* __restrict__ bo, const float* __restrict__ x,
                               float* __restrict__ ypre) {
  const int wave = (blockIdx.x * blockDim.x + threadIdx.x) >> 5;
  const int lane = threadIdx.x & 31;
  const int ntn = DD / 64;
  const int m0 = (wave / ntn) * 16;
  const int n0 = (wave % ntn) * 64;
  v8f acc[4] = {{}, {}, {}, {}};
  gemm_tile_bf16(ctx + (size_t)m0 * DD, DD, Wo + (size_t)n0 * DD, DD, DD, lane, acc);
  const int col = lane & 15, grp = lane >> 4;
#pragma unroll
  for (int t = 0; t < 4; ++t)
#pragma unroll
    for (int r = 0; r < 8; ++r) {
      const int m = m0 + r + 8 * grp;
      const int n = n0 + t * 16 + col;
      ypre[(size_t)m * DD + n] = acc[t][r] + bo[n] + x[(size_t)m * DD + n];
    }
}

// Row LayerNorm over 1024, write y to d_out.
__global__ void ln_kernel(const float* __restrict__ ypre, const float* __restrict__ g,
                          const float* __restrict__ bta, float* __restrict__ y) {
  __shared__ float r0[256];
  __shared__ float r1[256];
  const float* p = ypre + (size_t)blockIdx.x * DD;
  float* q = y + (size_t)blockIdx.x * DD;
  const int tid = threadIdx.x;
  float s = 0.f, s2 = 0.f;
  for (int i = tid; i < DD; i += 256) { float v = p[i]; s += v; s2 += v * v; }
  r0[tid] = s; r1[tid] = s2; __syncthreads();
  for (int st = 128; st > 0; st >>= 1) {
    if (tid < st) { r0[tid] += r0[tid + st]; r1[tid] += r1[tid + st]; }
    __syncthreads();
  }
  const float mu  = r0[0] * (1.0f / DD);
  const float var = r1[0] * (1.0f / DD) - mu * mu;
  const float rs  = rsqrtf(var + LN_EPS);
  for (int i = tid; i < DD; i += 256)
    q[i] = (p[i] - mu) * rs * g[i] + bta[i];
}

// ------------------------------- Launch ------------------------------------

extern "C" void kernel_launch(void* const* d_in, const int* in_sizes, int n_in,
                              void* d_out, int out_size, void* d_ws, size_t ws_size,
                              hipStream_t stream) {
  const float* x    = (const float*)d_in[0];
  const float* rst  = (const float*)d_in[1];
  const float* Wq   = (const float*)d_in[2];
  const float* bq   = (const float*)d_in[3];
  const float* Wk   = (const float*)d_in[4];
  const float* bk   = (const float*)d_in[5];
  const float* Wv   = (const float*)d_in[6];
  const float* bv   = (const float*)d_in[7];
  const float* Wr   = (const float*)d_in[8];
  const float* br   = (const float*)d_in[9];
  const float* Wo   = (const float*)d_in[10];
  const float* bo   = (const float*)d_in[11];
  const float* ln_g = (const float*)d_in[12];
  const float* ln_b = (const float*)d_in[13];

  float* y    = (float*)d_out;                       // [2,2048,1024]
  float* attn = y + (size_t)MM * DD;                 // [2,16,2048,2048]

  char* w = (char*)d_ws;
  auto carve = [&](size_t bytes) {
    char* p = w;
    w += (bytes + 255) & ~(size_t)255;
    return p;
  };
  bf16*  xb   = (bf16*)carve((size_t)MM * DD * 2);
  bf16*  Wqb  = (bf16*)carve((size_t)DD * DD * 2);
  bf16*  Wkb  = (bf16*)carve((size_t)DD * DD * 2);
  bf16*  Wvb  = (bf16*)carve((size_t)DD * DD * 2);
  bf16*  Wob  = (bf16*)carve((size_t)DD * DD * 2);
  bf16*  Qb   = (bf16*)carve((size_t)MM * DD * 2);
  bf16*  Kb   = (bf16*)carve((size_t)MM * DD * 2);
  bf16*  Vt   = (bf16*)carve((size_t)BB * DD * SS * 2);
  bf16*  ctx  = (bf16*)carve((size_t)MM * DD * 2);
  float* Rf   = (float*)carve((size_t)BB * DD * 4);
  float* qr   = (float*)carve((size_t)BB * HH * SS * 4);
  float* ypre = (float*)carve((size_t)MM * DD * 4);

  const int TB = 256;

  // 1. fp32 -> bf16 conversions
  cvt_kernel<<<(MM * DD + TB - 1) / TB, TB, 0, stream>>>(x, xb, MM * DD);
  cvt_kernel<<<(DD * DD + TB - 1) / TB, TB, 0, stream>>>(Wq, Wqb, DD * DD);
  cvt_kernel<<<(DD * DD + TB - 1) / TB, TB, 0, stream>>>(Wk, Wkb, DD * DD);
  cvt_kernel<<<(DD * DD + TB - 1) / TB, TB, 0, stream>>>(Wv, Wvb, DD * DD);
  cvt_kernel<<<(DD * DD + TB - 1) / TB, TB, 0, stream>>>(Wo, Wob, DD * DD);

  // 2. Q / K / V projections (WMMA)
  const int proj_blocks = (MM / 16) * (DD / 64) / 8;   // 512
  gemm_proj_kernel<<<proj_blocks, TB, 0, stream>>>(xb, Wqb, bq, Qb);
  gemm_proj_kernel<<<proj_blocks, TB, 0, stream>>>(xb, Wkb, bk, Kb);
  gemm_v_kernel  <<<proj_blocks, TB, 0, stream>>>(xb, Wvb, bv, Vt);

  // 3. reasoning projection + per-row score bias
  rproj_kernel<<<(BB * DD + TB - 1) / TB, TB, 0, stream>>>(rst, Wr, br, Rf);
  qr_kernel<<<(BB * HH * SS + TB - 1) / TB, TB, 0, stream>>>(Qb, Rf, qr);

  // 4. fused scores + softmax + attn write + context (WMMA)
  const int fused_blocks = BB * HH * (SS / 16) / 8;    // 512
  fused_attn_kernel<<<fused_blocks, TB, 0, stream>>>(Qb, Kb, Vt, qr, attn, ctx);

  // 5. output projection + residual (WMMA), then LayerNorm
  outproj_kernel<<<proj_blocks, TB, 0, stream>>>(ctx, Wob, bo, x, ypre);
  ln_kernel<<<MM, TB, 0, stream>>>(ypre, ln_g, ln_b, y);
}
